// SwinTransformerBlock3D_74663711474176
// MI455X (gfx1250) — matrix operand order repaired and checked by reading, not compiled, not run on hardware
//
#include <hip/hip_runtime.h>
#include <hip/hip_bf16.h>
#include <math.h>

typedef __attribute__((ext_vector_type(16))) _Float16 v16h;
typedef __attribute__((ext_vector_type(8)))  _Float16 v8h;
typedef __attribute__((ext_vector_type(8)))  float    v8f;
typedef __attribute__((ext_vector_type(4)))  unsigned u32x4;
typedef __attribute__((ext_vector_type(8)))  int      i32x8;
typedef __attribute__((ext_vector_type(4)))  int      i32x4;

union V16U { v16h v; v8h h8[2]; };
union U4H8 { uint4 u; _Float16 h[8]; };

// ---------------------------------------------------------------------------
// Geometry: Z=16,H=32,W=32, WS=4, SHIFT=2, C=384, HEADS=12, HEAD_DIM=32,
// N=64 tokens/window, 1024 windows, 65536 token rows.
// ---------------------------------------------------------------------------

__device__ __forceinline__ size_t remap_token(int row) {
    // (win, t) -> original flat token index (window reverse + roll(+2))
    int win = row >> 6, t = row & 63;
    int b  = win >> 8, wr = win & 255;
    int wz = wr >> 6, wy = (wr >> 3) & 7, wx = wr & 7;
    int lz = t >> 4,  ly = (t >> 2) & 3,  lx = t & 3;
    int z = (wz * 4 + lz + 2) & 15;
    int h = (wy * 4 + ly + 2) & 31;
    int w = (wx * 4 + lx + 2) & 31;
    return (size_t)b * 16384 + (size_t)z * 1024 + h * 32 + w;
}

// ---------------------------------------------------------------------------
// TDM: issue a 2D tensor_load_to_lds (f16 elements).
// D# group0: count=1, lds_addr, global_addr, type=2.
// D# group1: data_size=1(2B), tensor_dim0=rowLen, tensor_dim1=big,
//            tile_dim0 (k elems), tile_dim1 (rows), stride0=rowLen.
// ---------------------------------------------------------------------------
__device__ __forceinline__ unsigned lds_addr32(const void* p) {
    // generic pointers to LDS carry the LDS byte offset in the low 32 bits
    return (unsigned)(uintptr_t)p;
}

__device__ __forceinline__ void tdm_load_2d(unsigned lds_byte, const void* gaddr,
                                            unsigned tile_k, unsigned tile_rows,
                                            unsigned row_len) {
    unsigned long long ga = (unsigned long long)(uintptr_t)gaddr;
    u32x4 g0;
    g0[0] = 1u;                                   // count=1, user descriptor
    g0[1] = lds_byte;                             // lds_addr
    g0[2] = (unsigned)(ga & 0xffffffffu);         // global_addr[31:0]
    g0[3] = (unsigned)((ga >> 32) & 0x01ffffffu)  // global_addr[56:32]
          | 0x80000000u;                          // type=2 ("image")
    unsigned td0 = row_len;                       // tensor_dim0 (elements)
    unsigned td1 = 1u << 20;                      // tensor_dim1 (large, OOB slack)
    i32x8 g1;
    g1[0] = (int)(1u << 16);                      // data_size=1 (2 bytes)
    g1[1] = (int)((td0 & 0xffffu) << 16);         // tensor_dim0[15:0] @ bits 63:48
    g1[2] = (int)(((td0 >> 16) & 0xffffu) | ((td1 & 0xffffu) << 16));
    g1[3] = (int)(((td1 >> 16) & 0xffffu) | ((tile_k & 0xffffu) << 16)); // tile_dim0
    g1[4] = (int)(tile_rows & 0xffffu);           // tile_dim1 (tile_dim2=0)
    g1[5] = (int)row_len;                         // tensor_dim0_stride[31:0]
    g1[6] = 0;                                    // stride0[47:32]=0, stride1 lo=0
    g1[7] = 0;
    i32x4 z4 = {0, 0, 0, 0};
#if defined(__clang_major__) && (__clang_major__ >= 23)
    i32x8 z8 = {0, 0, 0, 0, 0, 0, 0, 0};
    __builtin_amdgcn_tensor_load_to_lds(g0, g1, z4, z4, z8, 0);
#else
    __builtin_amdgcn_tensor_load_to_lds(g0, g1, z4, z4, 0);
#endif
}

// ---------------------------------------------------------------------------
// float -> f16 weight conversion; transposed variant writes W^T [N][K]
// ---------------------------------------------------------------------------
__global__ void cvt_f16_t(const float* __restrict__ s, _Float16* __restrict__ d,
                          int Kdim, int Ndim) {
    int i = blockIdx.x * 256 + threadIdx.x;
    if (i < Kdim * Ndim) {
        int k = i / Ndim, n = i % Ndim;
        d[(size_t)n * Kdim + k] = (_Float16)s[i];
    }
}

// ---------------------------------------------------------------------------
// LayerNorm over C=384; remap=1 also does roll(-2)+window partition.
// ---------------------------------------------------------------------------
__global__ __launch_bounds__(128) void ln_win_kernel(
    const float* __restrict__ x, const float* __restrict__ g,
    const float* __restrict__ bvec, _Float16* __restrict__ out, int remap) {
    int r = blockIdx.x;
    int tid = threadIdx.x;
    size_t src = remap ? remap_token(r) : (size_t)r;
    const float* px = x + src * 384;
    float v0 = px[tid], v1 = px[tid + 128], v2 = px[tid + 256];
    __shared__ float s1[128], s2[128];
    s1[tid] = v0 + v1 + v2;
    s2[tid] = v0 * v0 + v1 * v1 + v2 * v2;
    __syncthreads();
    for (int st = 64; st > 0; st >>= 1) {
        if (tid < st) { s1[tid] += s1[tid + st]; s2[tid] += s2[tid + st]; }
        __syncthreads();
    }
    float mu  = s1[0] * (1.0f / 384.0f);
    float var = s2[0] * (1.0f / 384.0f) - mu * mu;
    float rs  = rsqrtf(var + 1e-5f);
    _Float16* po = out + (size_t)r * 384;
    po[tid]       = (_Float16)((v0 - mu) * rs * g[tid]       + bvec[tid]);
    po[tid + 128] = (_Float16)((v1 - mu) * rs * g[tid + 128] + bvec[tid + 128]);
    po[tid + 256] = (_Float16)((v2 - mu) * rs * g[tid + 256] + bvec[tid + 256]);
}

// ---------------------------------------------------------------------------
// f16 WMMA GEMM, TDM-staged, double-buffered. out = epi(A[M,K] x B^T + bias)
//   Bw is pre-transposed: Bw[n][k], row length K.
//   mode 0: f16 out (QKV)   mode 1: f16 GELU out (fc1)
//   mode 2: f32 out + residual, optional window-reverse remap (proj/fc2)
// Block 256 threads = 8 waves (2 row x 4 col), block tile 64x128,
// wave tile 32x32 (2x2 WMMA tiles), K-step 32.
// ---------------------------------------------------------------------------
__global__ __launch_bounds__(256) void wmma_gemm(
    const _Float16* __restrict__ A, const _Float16* __restrict__ Bw,
    const float* __restrict__ bias, int N, int K, int mode,
    _Float16* __restrict__ outH, float* __restrict__ outF,
    const float* __restrict__ skip, int remap) {
    __shared__ _Float16 As[2][64 * 32];    // [m][k] per buffer
    __shared__ _Float16 Bs[2][128 * 32];   // [n][k] per buffer
    __shared__ float    Cst[64 * 128];     // epilogue staging (f32 or f16 alias)

    int tid  = threadIdx.x;
    int wave = tid >> 5, lane = tid & 31;
    int rowBase = blockIdx.y * 64, colBase = blockIdx.x * 128;
    int wr = wave >> 2;          // 0..1 : 32-row strip
    int wc = wave & 3;           // 0..3 : 32-col strip
    int lm  = lane & 15;
    int akb = (lane < 16) ? 0 : 16;   // A fragment k-base
    int bkb = (lane < 16) ? 0 : 16;   // B fragment k-base
    int mhi = (lane < 16) ? 0 : 8;    // C/D row offset (hi half-wave)
    v8f acc[2][2] = {{{}, {}}, {{}, {}}};

    const int nsteps = K >> 5;
    if (wave == 0) {
        tdm_load_2d(lds_addr32(As[0]), A  + (size_t)rowBase * K, 32, 64,  (unsigned)K);
        tdm_load_2d(lds_addr32(Bs[0]), Bw + (size_t)colBase * K, 32, 128, (unsigned)K);
        __builtin_amdgcn_s_wait_tensorcnt(0);
    }
    __syncthreads();

    for (int i = 0; i < nsteps; ++i) {
        int cur = i & 1;
        if (wave == 0 && (i + 1) < nsteps) {   // prefetch next tiles via TDM
            int ks = (i + 1) << 5;
            tdm_load_2d(lds_addr32(As[cur ^ 1]), A  + (size_t)rowBase * K + ks, 32, 64,  (unsigned)K);
            tdm_load_2d(lds_addr32(Bs[cur ^ 1]), Bw + (size_t)colBase * K + ks, 32, 128, (unsigned)K);
        }
        const _Float16* as = As[cur];
        const _Float16* bs = Bs[cur];
        V16U af[2];
        #pragma unroll
        for (int rt = 0; rt < 2; ++rt) {
            int mrow = 32 * wr + 16 * rt + lm;
            af[rt].h8[0] = *(const v8h*)(as + mrow * 32 + akb);
            af[rt].h8[1] = *(const v8h*)(as + mrow * 32 + 8 + akb);
        }
        #pragma unroll
        for (int ct = 0; ct < 2; ++ct) {
            int nrow = 32 * wc + 16 * ct + lm;
            v16h bf = *(const v16h*)(bs + nrow * 32 + bkb);
            #pragma unroll
            for (int rt = 0; rt < 2; ++rt)
                acc[rt][ct] = __builtin_amdgcn_wmma_f32_16x16x32_f16(
                    false, af[rt].v, false, bf, (short)0, acc[rt][ct], false, false);
        }
        if (wave == 0) __builtin_amdgcn_s_wait_tensorcnt(0);
        __syncthreads();
    }

    // ---- epilogue: stage tile in LDS, then coalesced global writes ----
    if (mode <= 1) {
        _Float16* CstH = (_Float16*)Cst;
        #pragma unroll
        for (int rt = 0; rt < 2; ++rt)
            #pragma unroll
            for (int ct = 0; ct < 2; ++ct) {
                int n = 32 * wc + 16 * ct + lm;
                float bv = bias[colBase + n];
                #pragma unroll
                for (int v = 0; v < 8; ++v) {
                    int m = 32 * wr + 16 * rt + v + mhi;
                    float val = acc[rt][ct][v] + bv;
                    if (mode == 1) val = 0.5f * val * (1.0f + erff(val * 0.70710678f));
                    CstH[m * 128 + n] = (_Float16)val;
                }
            }
        __syncthreads();
        #pragma unroll
        for (int j = 0; j < 4; ++j) {      // 64x128 f16 = 1024 uint4
            int u = j * 256 + tid;
            int row = u >> 4, c8 = (u & 15) * 8;
            *(uint4*)(outH + (size_t)(rowBase + row) * N + colBase + c8) =
                *(const uint4*)(CstH + row * 128 + c8);
        }
    } else {
        #pragma unroll
        for (int rt = 0; rt < 2; ++rt)
            #pragma unroll
            for (int ct = 0; ct < 2; ++ct) {
                int n = 32 * wc + 16 * ct + lm;
                float bv = bias[colBase + n];
                #pragma unroll
                for (int v = 0; v < 8; ++v) {
                    int m = 32 * wr + 16 * rt + v + mhi;
                    Cst[m * 128 + n] = acc[rt][ct][v] + bv;
                }
            }
        __syncthreads();
        #pragma unroll
        for (int j = 0; j < 8; ++j) {      // 64x128 f32 = 2048 float4
            int u = j * 256 + tid;
            int row = u >> 5, c4 = (u & 31) * 4;
            size_t dst = remap ? remap_token(rowBase + row) : (size_t)(rowBase + row);
            size_t idx = dst * (size_t)N + colBase + c4;
            float4 cv = *(const float4*)(Cst + row * 128 + c4);
            float4 sv = *(const float4*)(skip + idx);
            float4 ov = make_float4(cv.x + sv.x, cv.y + sv.y, cv.z + sv.z, cv.w + sv.w);
            *(float4*)(outF + idx) = ov;
        }
    }
}

// ---------------------------------------------------------------------------
// Windowed attention: one block per (window, head). 128 threads = 4 waves.
// S = (Q K^T)*scale + rel_bias + shift_mask ; softmax ; O = P V.
// qkv layout: [65536 rows, 1152] with col = which*384 + head*32 + d.
// ---------------------------------------------------------------------------
__global__ __launch_bounds__(128) void attn_kernel(
    const _Float16* __restrict__ qkv, const float* __restrict__ rel_bias,
    _Float16* __restrict__ outH) {
    __shared__ _Float16 Qs[64 * 32];   // [m][d]
    __shared__ _Float16 Ks[64 * 32];   // [n][d]
    __shared__ _Float16 Vt[32 * 64];   // [d][token]
    __shared__ float    Sf[64 * 64];
    __shared__ _Float16 Ps[64 * 64];
    int head = blockIdx.x;
    int win  = blockIdx.y;
    int tid  = threadIdx.x;
    int wave = tid >> 5, lane = tid & 31;

    { // stage Q, K, V (16 halves per thread per matrix)
        int row = tid >> 1, part = tid & 1;
        size_t base = (size_t)(win * 64 + row) * 1152 + head * 32 + part * 16;
        *(uint4*)(Qs + row * 32 + part * 16)     = *(const uint4*)(qkv + base);
        *(uint4*)(Qs + row * 32 + part * 16 + 8) = *(const uint4*)(qkv + base + 8);
        *(uint4*)(Ks + row * 32 + part * 16)     = *(const uint4*)(qkv + base + 384);
        *(uint4*)(Ks + row * 32 + part * 16 + 8) = *(const uint4*)(qkv + base + 392);
        U4H8 a, b;
        a.u = *(const uint4*)(qkv + base + 768);
        b.u = *(const uint4*)(qkv + base + 776);
        #pragma unroll
        for (int j = 0; j < 8; ++j) {
            Vt[(part * 16 + j) * 64 + row]     = a.h[j];
            Vt[(part * 16 + 8 + j) * 64 + row] = b.h[j];
        }
    }
    __syncthreads();

    int lm  = lane & 15;
    int akb = (lane < 16) ? 0 : 16;
    int bkb = (lane < 16) ? 0 : 16;
    int mhi = (lane < 16) ? 0 : 8;
    int wrm = win & 255;
    int wz = wrm >> 6, wy = (wrm >> 3) & 7, wx = wrm & 7;

    V16U qf;
    qf.h8[0] = *(const v8h*)(Qs + (16 * wave + lm) * 32 + akb);
    qf.h8[1] = *(const v8h*)(Qs + (16 * wave + lm) * 32 + 8 + akb);
    #pragma unroll
    for (int c = 0; c < 4; ++c) {
        v16h kf = *(const v16h*)(Ks + (16 * c + lm) * 32 + bkb);
        v8f s = {};
        s = __builtin_amdgcn_wmma_f32_16x16x32_f16(false, qf.v, false, kf,
                                                   (short)0, s, false, false);
        int n  = 16 * c + lm;
        int nz = n >> 4, ny = (n >> 2) & 3, nx = n & 3;
        int catn = ((wz == 3) ? ((nz < 2) ? 1 : 2) : 0) * 9 +
                   ((wy == 7) ? ((ny < 2) ? 1 : 2) : 0) * 3 +
                   ((wx == 7) ? ((nx < 2) ? 1 : 2) : 0);
        #pragma unroll
        for (int v = 0; v < 8; ++v) {
            int m  = 16 * wave + v + mhi;
            int mz = m >> 4, my = (m >> 2) & 3, mx = m & 3;
            int idx = (mz - nz + 3) * 49 + (my - ny + 3) * 7 + (mx - nx + 3);
            float val = s[v] * 0.17677669529663687f + rel_bias[idx * 12 + head];
            int catm = ((wz == 3) ? ((mz < 2) ? 1 : 2) : 0) * 9 +
                       ((wy == 7) ? ((my < 2) ? 1 : 2) : 0) * 3 +
                       ((wx == 7) ? ((mx < 2) ? 1 : 2) : 0);
            if (catm != catn) val -= 100.0f;
            Sf[m * 64 + n] = val;
        }
    }
    __syncthreads();

    if (tid < 64) {
        float mx = -1e30f;
        for (int j = 0; j < 64; ++j) mx = fmaxf(mx, Sf[tid * 64 + j]);
        float sum = 0.0f;
        for (int j = 0; j < 64; ++j) sum += expf(Sf[tid * 64 + j] - mx);
        float inv = 1.0f / sum;
        for (int j = 0; j < 64; ++j)
            Ps[tid * 64 + j] = (_Float16)(expf(Sf[tid * 64 + j] - mx) * inv);
    }
    __syncthreads();

    v8f o0 = {}, o1 = {};
    #pragma unroll
    for (int ksi = 0; ksi < 2; ++ksi) {
        V16U pf;
        pf.h8[0] = *(const v8h*)(Ps + (16 * wave + lm) * 64 + 32 * ksi + akb);
        pf.h8[1] = *(const v8h*)(Ps + (16 * wave + lm) * 64 + 32 * ksi + 8 + akb);
        v16h vb0 = *(const v16h*)(Vt + lm * 64        + 32 * ksi + bkb);
        v16h vb1 = *(const v16h*)(Vt + (16 + lm) * 64 + 32 * ksi + bkb);
        o0 = __builtin_amdgcn_wmma_f32_16x16x32_f16(false, pf.v, false, vb0,
                                                    (short)0, o0, false, false);
        o1 = __builtin_amdgcn_wmma_f32_16x16x32_f16(false, pf.v, false, vb1,
                                                    (short)0, o1, false, false);
    }
    #pragma unroll
    for (int v = 0; v < 8; ++v) {
        int m = 16 * wave + v + mhi;
        size_t base = (size_t)(win * 64 + m) * 384 + head * 32;
        outH[base + lm]      = (_Float16)o0[v];
        outH[base + 16 + lm] = (_Float16)o1[v];
    }
}

// ---------------------------------------------------------------------------
// Launch
// ---------------------------------------------------------------------------
extern "C" void kernel_launch(void* const* d_in, const int* in_sizes, int n_in,
                              void* d_out, int out_size, void* d_ws, size_t ws_size,
                              hipStream_t stream) {
    const float* x        = (const float*)d_in[0];
    const float* n1g      = (const float*)d_in[1];
    const float* n1b      = (const float*)d_in[2];
    const float* qkv_w    = (const float*)d_in[3];
    const float* qkv_b    = (const float*)d_in[4];
    const float* proj_w   = (const float*)d_in[5];
    const float* proj_b   = (const float*)d_in[6];
    const float* rel_bias = (const float*)d_in[7];
    const float* n2g      = (const float*)d_in[8];
    const float* n2b      = (const float*)d_in[9];
    const float* fc1_w    = (const float*)d_in[10];
    const float* fc1_b    = (const float*)d_in[11];
    const float* fc2_w    = (const float*)d_in[12];
    const float* fc2_b    = (const float*)d_in[13];
    float* out = (float*)d_out;

    char* ws = (char*)d_ws;
    size_t off = 0;
    auto take = [&](size_t bytes) -> char* {
        char* p = ws + off;
        off += (bytes + 255) & ~(size_t)255;
        return p;
    };
    const size_t M = 65536;
    _Float16* wqkvT = (_Float16*)take((size_t)1152 * 384 * 2);  // [N][K]
    _Float16* wprjT = (_Float16*)take((size_t)384 * 384 * 2);
    _Float16* wfc1T = (_Float16*)take((size_t)1536 * 384 * 2);
    _Float16* wfc2T = (_Float16*)take((size_t)384 * 1536 * 2);
    _Float16* winA  = (_Float16*)take(M * 384 * 2);   // LN1 out; reused as attn out
    _Float16* qkvh  = (_Float16*)take(M * 1152 * 2);
    float*    x2    = (float*)   take(M * 384 * 4);
    _Float16* h2    = (_Float16*)take(M * 384 * 2);
    _Float16* mlp1  = (_Float16*)take(M * 1536 * 2);

    // weights -> f16, transposed to [N][K]
    cvt_f16_t<<<(384 * 1152 + 255) / 256, 256, 0, stream>>>(qkv_w, wqkvT, 384, 1152);
    cvt_f16_t<<<(384 * 384 + 255) / 256, 256, 0, stream>>>(proj_w, wprjT, 384, 384);
    cvt_f16_t<<<(384 * 1536 + 255) / 256, 256, 0, stream>>>(fc1_w, wfc1T, 384, 1536);
    cvt_f16_t<<<(1536 * 384 + 255) / 256, 256, 0, stream>>>(fc2_w, wfc2T, 1536, 384);

    // LN1 + roll(-2) + window partition -> f16
    ln_win_kernel<<<M, 128, 0, stream>>>(x, n1g, n1b, winA, 1);
    // QKV GEMM
    wmma_gemm<<<dim3(1152 / 128, M / 64), 256, 0, stream>>>(
        winA, wqkvT, qkv_b, 1152, 384, 0, qkvh, nullptr, nullptr, 0);
    // attention (winA reused as attn output)
    attn_kernel<<<dim3(12, 1024), 128, 0, stream>>>(qkvh, rel_bias, winA);
    // proj GEMM + window reverse + roll(+2) + residual -> x2
    wmma_gemm<<<dim3(384 / 128, M / 64), 256, 0, stream>>>(
        winA, wprjT, proj_b, 384, 384, 2, nullptr, x2, x, 1);
    // LN2
    ln_win_kernel<<<M, 128, 0, stream>>>(x2, n2g, n2b, h2, 0);
    // fc1 + GELU
    wmma_gemm<<<dim3(1536 / 128, M / 64), 256, 0, stream>>>(
        h2, wfc1T, fc1_b, 1536, 384, 1, mlp1, nullptr, nullptr, 0);
    // fc2 + residual -> d_out
    wmma_gemm<<<dim3(384 / 128, M / 64), 256, 0, stream>>>(
        mlp1, wfc2T, fc2_b, 384, 1536, 2, nullptr, out, x2, 0);
}
